// PrototypeLayer_56255481643513
// MI455X (gfx1250) — matrix operand order repaired and checked by reading, not compiled, not run on hardware
//
#include <hip/hip_runtime.h>
#include <stdint.h>

typedef __attribute__((ext_vector_type(16))) __bf16        v16bf;
typedef __attribute__((ext_vector_type(8)))  float         v8f;
typedef __attribute__((ext_vector_type(2)))  unsigned int  u32x2;
typedef __attribute__((ext_vector_type(4)))  unsigned int  u32x4;
typedef __attribute__((ext_vector_type(8)))  unsigned int  u32x8;

static constexpr int HDIM  = 1024;    // hidden dim (K)
static constexpr int NPROT = 512;     // prototypes (N)
static constexpr int NTOK  = 65536;   // 32*2048 tokens (M)
static constexpr int BM    = 32;      // tokens per block (2 M-tiles)
static constexpr int KC    = 32;      // bf16 WMMA K depth
static constexpr int MT    = BM / 16; // M-tiles per block (=2)

__device__ __forceinline__ unsigned short f32_to_bf16(float f) {
    unsigned u = __builtin_bit_cast(unsigned, f);
    u += 0x7FFFu + ((u >> 16) & 1u);          // round-to-nearest-even
    return (unsigned short)(u >> 16);
}

// JAX threefry2x32 with key(42) = (0,42); counts = iota(2^25) split in half.
// Returns gumbel = -log(-log(uniform(EPS,1))).
__device__ __forceinline__ float gumbel_noise(unsigned e) {
    const unsigned k0 = 0u, k1 = 42u;
    const unsigned k2 = 0x1BD11BDAu ^ k0 ^ k1;
    const unsigned HALF = 1u << 24;           // (2^25)/2 pairs
    unsigned i  = e & (HALF - 1u);
    unsigned x0 = i + k0;
    unsigned x1 = (i + HALF) + k1;
#define TFR(r) { x0 += x1; x1 = (x1 << (r)) | (x1 >> (32 - (r))); x1 ^= x0; }
    TFR(13) TFR(15) TFR(26) TFR(6)   x0 += k1; x1 += k2 + 1u;
    TFR(17) TFR(29) TFR(16) TFR(24)  x0 += k2; x1 += k0 + 2u;
    TFR(13) TFR(15) TFR(26) TFR(6)   x0 += k0; x1 += k1 + 3u;
    TFR(17) TFR(29) TFR(16) TFR(24)  x0 += k1; x1 += k2 + 4u;
    TFR(13) TFR(15) TFR(26) TFR(6)   x0 += k2; x1 += k0 + 5u;
#undef TFR
    unsigned bits = (e & HALF) ? x1 : x0;
    float f = __builtin_bit_cast(float, (bits >> 9) | 0x3F800000u) - 1.0f;
    float u = f * (1.0f - 1e-20f) + 1e-20f;
    u = fmaxf(u, 1e-20f);
    return -__logf(-__logf(u));
}

// ---- Kernel A: prototypes fp32 -> bf16 (for WMMA B operand) + ||p||^2 ----
__global__ __launch_bounds__(256) void proto_prep_kernel(
    const float* __restrict__ protos,
    unsigned short* __restrict__ pbf,
    float* __restrict__ pnorm2)
{
    __shared__ float red[256];
    const int row = blockIdx.x;
    const int t   = threadIdx.x;
    const float4 v = *(const float4*)(protos + (size_t)row * HDIM + t * 4);
    u32x2 packed;
    packed.x = (unsigned)f32_to_bf16(v.x) | ((unsigned)f32_to_bf16(v.y) << 16);
    packed.y = (unsigned)f32_to_bf16(v.z) | ((unsigned)f32_to_bf16(v.w) << 16);
    *(u32x2*)(pbf + (size_t)row * HDIM + t * 4) = packed;
    red[t] = v.x * v.x + v.y * v.y + v.z * v.z + v.w * v.w;
    __syncthreads();
    for (int s = 128; s > 0; s >>= 1) {
        if (t < s) red[t] += red[t + s];
        __syncthreads();
    }
    if (t == 0) pnorm2[row] = red[0];
}

// ---- Kernel B: fused distance-GEMM + gumbel-argmax + gather/residual ----
__global__ __launch_bounds__(256, 1) void proto_assign_kernel(
    const float* __restrict__ x,
    const float* __restrict__ protos,
    const unsigned short* __restrict__ pbf,
    const float* __restrict__ pnorm2,
    float* __restrict__ out_proto,
    float* __restrict__ out_res)
{
    __shared__ unsigned short lds_x[BM * KC];        // 2 KB  (A tile, bf16)
    __shared__ unsigned short lds_p[NPROT * KC];     // 32 KB (B tile, bf16)
    __shared__ float lds_xn[BM];                     // ||x||^2 per token
    __shared__ float lds_pn[NPROT];                  // ||p||^2 cache
    __shared__ unsigned long long lds_best[BM];      // packed argmax keys

    const int tid   = threadIdx.x;
    const int wave  = tid >> 5;        // 8 waves, wave w owns N in [64w, 64w+64)
    const int lane  = tid & 31;
    const int laneN = lane & 15;
    const int hlf   = lane >> 4;       // half-wave select
    const int tokBase = blockIdx.x * BM;

    if (tid < BM) { lds_xn[tid] = 0.0f; lds_best[tid] = 0ull; }
    lds_pn[tid]       = pnorm2[tid];
    lds_pn[tid + 256] = pnorm2[tid + 256];

    const v8f vzero = {0, 0, 0, 0, 0, 0, 0, 0};
    v8f acc[MT][4];
#pragma unroll
    for (int mt = 0; mt < MT; ++mt)
#pragma unroll
        for (int nt = 0; nt < 4; ++nt) acc[mt][nt] = vzero;

    const int xr = tid >> 3;           // token row within block (0..31)
    const int xc = (tid & 7) * 4;      // 4 contiguous k per thread
    float xsq = 0.0f;
    const float* xrow = x + (size_t)(tokBase + xr) * HDIM + xc;

    for (int kc = 0; kc < HDIM; kc += KC) {
        // stage x tile fp32->bf16, accumulate |x|^2 partials
        float4 a0 = *(const float4*)(xrow + kc);
        if (kc + KC < HDIM) __builtin_prefetch(xrow + kc + KC, 0, 3);
        xsq += a0.x * a0.x + a0.y * a0.y + a0.z * a0.z + a0.w * a0.w;
        u32x2 xpk;
        xpk.x = (unsigned)f32_to_bf16(a0.x) | ((unsigned)f32_to_bf16(a0.y) << 16);
        xpk.y = (unsigned)f32_to_bf16(a0.z) | ((unsigned)f32_to_bf16(a0.w) << 16);
        *(u32x2*)&lds_x[xr * KC + xc] = xpk;

        // stage prototype tile (already bf16 in ws): 512 rows x 32 k
#pragma unroll
        for (int g = 0; g < 8; ++g) {
            const int pr = g * 64 + (tid >> 2);
            const u32x4* src = (const u32x4*)(pbf + (size_t)pr * HDIM + kc);
            u32x4 pv = src[tid & 3];
            *(u32x4*)&lds_p[pr * KC + (tid & 3) * 8] = pv;
        }
        __syncthreads();

        // A fragments: lanes 0-15 K=[0..7,16..23], lanes 16-31 K=[8..15,24..31]
        const int kA = hlf * 8;
        v16bf afr[MT];
#pragma unroll
        for (int mt = 0; mt < MT; ++mt) {
            u32x4 lo = *(const u32x4*)&lds_x[(mt * 16 + laneN) * KC + kA];
            u32x4 hi = *(const u32x4*)&lds_x[(mt * 16 + laneN) * KC + kA + 16];
            u32x8 f = {lo.x, lo.y, lo.z, lo.w, hi.x, hi.y, hi.z, hi.w};
            afr[mt] = __builtin_bit_cast(v16bf, f);
        }
        // B fragments: lane = column n, lanes 0-15 K=[0..7,16..23], lanes 16-31 K=[8..15,24..31]
        const int kB = hlf * 8;
        v16bf bfr[4];
#pragma unroll
        for (int nt = 0; nt < 4; ++nt) {
            const int pr = wave * 64 + nt * 16 + laneN;
            u32x4 lo = *(const u32x4*)&lds_p[pr * KC + kB];
            u32x4 hi = *(const u32x4*)&lds_p[pr * KC + kB + 16];
            u32x8 f = {lo.x, lo.y, lo.z, lo.w, hi.x, hi.y, hi.z, hi.w};
            bfr[nt] = __builtin_bit_cast(v16bf, f);
        }
#pragma unroll
        for (int mt = 0; mt < MT; ++mt)
#pragma unroll
            for (int nt = 0; nt < 4; ++nt)
                acc[mt][nt] = __builtin_amdgcn_wmma_f32_16x16x32_bf16(
                    false, afr[mt], false, bfr[nt], (short)0, acc[mt][nt],
                    false, false);
        __syncthreads();
    }

    atomicAdd(&lds_xn[xr], xsq);
    __syncthreads();

    // perturbed-logit argmax.  C layout: vgpr r -> M = r (+8 upper half), N = laneN
#pragma unroll
    for (int mt = 0; mt < MT; ++mt) {
#pragma unroll
        for (int r = 0; r < 8; ++r) {
            const int tl = mt * 16 + r + hlf * 8;
            const unsigned tokG = (unsigned)(tokBase + tl);
            const float xn = lds_xn[tl];
            unsigned long long best = 0ull;
#pragma unroll
            for (int nt = 0; nt < 4; ++nt) {
                const int n = wave * 64 + nt * 16 + laneN;
                const float dot = acc[mt][nt][r];
                float d2 = xn + lds_pn[n] - 2.0f * dot;
                d2 = d2 > 0.0f ? d2 : 0.0f;
                const float dist = sqrtf(d2) + 1e-20f;
                const float s = -__logf(dist)
                              + gumbel_noise(tokG * (unsigned)NPROT + (unsigned)n);
                unsigned u = __builtin_bit_cast(unsigned, s);
                u = (u & 0x80000000u) ? ~u : (u | 0x80000000u);  // order-preserving
                const unsigned long long key =
                    ((unsigned long long)u << 32) | (unsigned)(~n); // ties: min n
                best = key > best ? key : best;
            }
#pragma unroll
            for (int off = 1; off < 16; off <<= 1) {
                const unsigned long long o = __shfl_xor(best, off, 16);
                best = o > best ? o : best;
            }
            if (laneN == 0) atomicMax(&lds_best[tl], best);
        }
    }
    __syncthreads();

    // gather winning prototype (exact fp32), write proto + residual
    for (int i = 0; i < BM / 8; ++i) {
        const int tl = wave * (BM / 8) + i;
        const size_t tokG = (size_t)tokBase + tl;
        const unsigned n = (~(unsigned)lds_best[tl]) & (NPROT - 1);
        const float4* xp = (const float4*)(x + tokG * HDIM);
        const float4* pp = (const float4*)(protos + (size_t)n * HDIM);
        float4* op  = (float4*)(out_proto + tokG * HDIM);
        float4* orr = (float4*)(out_res + tokG * HDIM);
#pragma unroll
        for (int c = lane; c < HDIM / 4; c += 32) {
            float4 xv = xp[c], pv = pp[c];
            op[c] = pv;
            float4 rv = {xv.x - pv.x, xv.y - pv.y, xv.z - pv.z, xv.w - pv.w};
            orr[c] = rv;
        }
    }
}

extern "C" void kernel_launch(void* const* d_in, const int* in_sizes, int n_in,
                              void* d_out, int out_size, void* d_ws, size_t ws_size,
                              hipStream_t stream) {
    const float* x      = (const float*)d_in[0];   // [32,2048,1024] fp32
    const float* protos = (const float*)d_in[1];   // [512,1024]    fp32

    unsigned short* pbf = (unsigned short*)d_ws;                 // 1 MiB bf16 protos
    float* pnorm2 = (float*)((char*)d_ws +
                             (size_t)NPROT * HDIM * sizeof(unsigned short));

    float* out_proto = (float*)d_out;                            // [65536,1024]
    float* out_res   = out_proto + (size_t)NTOK * HDIM;          // [65536,1024]

    hipLaunchKernelGGL(proto_prep_kernel, dim3(NPROT), dim3(256), 0, stream,
                       protos, pbf, pnorm2);
    hipLaunchKernelGGL(proto_assign_kernel, dim3(NTOK / BM), dim3(256), 0, stream,
                       x, protos, pbf, pnorm2, out_proto, out_res);
}
